// LocalAttention_41051297415674
// MI455X (gfx1250) — compile-verified, matrix operand order checked by reading, not run on hardware
//
#include <hip/hip_runtime.h>

#define EDIM  256
#define BATCH 2
#define SEQ   2048
#define TOFF  128          // left pad of K / V^T along t
#define SP    2368         // 128 + 2048 + 192 padded t extent
#define NEG   (-1e30f)

typedef __attribute__((ext_vector_type(16))) __bf16 v16bf;
typedef __attribute__((ext_vector_type(8)))  __bf16 v8bf;
typedef __attribute__((ext_vector_type(8)))  float  v8f;

// Load a 16-element bf16 WMMA fragment slice for one lane:
// elements [0..7] at p, elements [8..15] at p+16 (per CDNA5 16-bit A/B layout).
__device__ __forceinline__ v16bf load_frag(const __bf16* p) {
  v8bf lo = *(const v8bf*)(p);
  v8bf hi = *(const v8bf*)(p + 16);
  v16bf r;
#pragma unroll
  for (int i = 0; i < 8; ++i) { r[i] = lo[i]; r[i + 8] = hi[i]; }
  return r;
}

__global__ void zero_ws(uint32_t* p, int n) {
  int i = blockIdx.x * blockDim.x + threadIdx.x;
  int stride = gridDim.x * blockDim.x;
  for (; i < n; i += stride) p[i] = 0u;
}

__global__ void cvt_bf16(const float* __restrict__ src, __bf16* __restrict__ dst, int n) {
  int i = blockIdx.x * blockDim.x + threadIdx.x;
  int stride = gridDim.x * blockDim.x;
  for (; i < n; i += stride) dst[i] = (__bf16)src[i];
}

// y = x @ W^T + b for q/k/v.  One wave -> one 16x16 tile of one projection.
// waves: rt(256) x ct(16) x p(3) = 12288
__global__ __launch_bounds__(256) void qkv_proj(
    const __bf16* __restrict__ xb, const __bf16* __restrict__ wb,
    const float* __restrict__ bq, const float* __restrict__ bk,
    const float* __restrict__ bv,
    __bf16* __restrict__ qb, __bf16* __restrict__ kbuf, __bf16* __restrict__ vtb) {
  const int wave = (blockIdx.x * blockDim.x + threadIdx.x) >> 5;
  const int lane = threadIdx.x & 31;
  const int half = lane >> 4, l15 = lane & 15;
  const int rt = wave & 255;
  const int ct = (wave >> 8) & 15;
  const int p  = wave >> 12;

  const __bf16* w    = wb + (size_t)p * (EDIM * EDIM);
  const float*  bias = (p == 0) ? bq : (p == 1) ? bk : bv;

  const __bf16* arow = xb + (size_t)(rt * 16 + l15) * EDIM;
  const __bf16* brow = w  + (size_t)(ct * 16 + l15) * EDIM;

  v8f acc = {};
#pragma unroll
  for (int ks = 0; ks < 8; ++ks) {
    int eb = ks * 32 + half * 8;
    v16bf a = load_frag(arow + eb);
    v16bf b = load_frag(brow + eb);
    acc = __builtin_amdgcn_wmma_f32_16x16x32_bf16(false, a, false, b,
                                                  (short)0, acc, false, false);
  }
  const float bval = bias[ct * 16 + l15];
  const int r0 = rt * 16 + half * 8;   // D layout: VGPR j -> row j (+8 for hi half)
  const int c  = ct * 16 + l15;        // lane -> column
#pragma unroll
  for (int j = 0; j < 8; ++j) {
    int r = r0 + j;
    __bf16 val = (__bf16)(acc[j] + bval);
    if (p == 0) {
      qb[(size_t)r * EDIM + c] = val;
    } else {
      int b = r >> 11, s = r & (SEQ - 1);
      if (p == 1) kbuf[((size_t)(b * SP + TOFF + s)) * EDIM + c] = val;
      else        vtb [((size_t)(b * EDIM + c)) * SP + TOFF + s] = val;
    }
  }
}

// Fused sliding-window attention: one wave per 16-row s-tile.
// waves: b(2) x s-tile(128) = 256 -> 32 blocks x 8 waves
__global__ __launch_bounds__(256) void local_attn(
    const __bf16* __restrict__ qb, const __bf16* __restrict__ kbuf,
    const __bf16* __restrict__ vtb, float* __restrict__ out) {
  __shared__ __align__(16) __bf16 pbuf[8][16 * 32];
  const int wslot = threadIdx.x >> 5;
  const int wave  = (blockIdx.x * blockDim.x + threadIdx.x) >> 5;
  const int lane  = threadIdx.x & 31;
  const int half  = lane >> 4, l15 = lane & 15;
  const int bb = wave >> 7;
  const int S0 = (wave & 127) * 16;

  const __bf16* qrow = qb + ((size_t)(bb * SEQ + S0 + l15)) * EDIM;

  v8f o[16];
  v8f z = {};
#pragma unroll
  for (int n = 0; n < 16; ++n) o[n] = z;
  float m[8], ssum[8];
#pragma unroll
  for (int j = 0; j < 8; ++j) { m[j] = NEG; ssum[j] = 0.0f; }

  for (int chunk = 0; chunk < 9; ++chunk) {
    const int t0 = S0 - 128 + chunk * 32;

    // ---- scores: E[16 x 32] = Q x K^T (zero-padded K handles seq edges) ----
    const __bf16* klo = kbuf + ((size_t)(bb * SP + TOFF + t0 + l15)) * EDIM;
    const __bf16* khi = klo + (size_t)16 * EDIM;
    __builtin_prefetch(klo + (size_t)32 * EDIM, 0, 0);   // next chunk's K rows
    v8f e0 = z, e1 = z;
#pragma unroll
    for (int ks = 0; ks < 8; ++ks) {
      int eb = ks * 32 + half * 8;
      v16bf a  = load_frag(qrow + eb);
      v16bf b0 = load_frag(klo + eb);
      v16bf b1 = load_frag(khi + eb);
      e0 = __builtin_amdgcn_wmma_f32_16x16x32_bf16(false, a, false, b0,
                                                   (short)0, e0, false, false);
      e1 = __builtin_amdgcn_wmma_f32_16x16x32_bf16(false, a, false, b1,
                                                   (short)0, e1, false, false);
    }

    // ---- window mask + online softmax (rows split across wave halves) ----
    const int dbase = chunk * 32 - 128;   // (t - s) = dbase + tt - r
#pragma unroll
    for (int j = 0; j < 8; ++j) {
      const int r = half * 8 + j;
      const int dlo = dbase + l15 - r;
      const int dhi = dlo + 16;
      const bool wl = (dlo >= -128) && (dlo <= 128);
      const bool wh = (dhi >= -128) && (dhi <= 128);
      float elo = wl ? e0[j] : NEG;
      float ehi = wh ? e1[j] : NEG;

      float mx = fmaxf(elo, ehi);
      mx = fmaxf(mx, __shfl_xor(mx, 1, 32));
      mx = fmaxf(mx, __shfl_xor(mx, 2, 32));
      mx = fmaxf(mx, __shfl_xor(mx, 4, 32));
      mx = fmaxf(mx, __shfl_xor(mx, 8, 32));

      float mnew  = fmaxf(m[j], mx);
      float scale = __expf(m[j] - mnew);
      float plo = __expf(elo - mnew);   // underflows to 0 when masked
      float phi = __expf(ehi - mnew);

      float rs = plo + phi;
      rs += __shfl_xor(rs, 1, 32);
      rs += __shfl_xor(rs, 2, 32);
      rs += __shfl_xor(rs, 4, 32);
      rs += __shfl_xor(rs, 8, 32);

      ssum[j] = ssum[j] * scale + rs;
      m[j] = mnew;

      pbuf[wslot][r * 32 + l15]      = (__bf16)plo;
      pbuf[wslot][r * 32 + l15 + 16] = (__bf16)phi;
#pragma unroll
      for (int n = 0; n < 16; ++n) o[n][j] *= scale;
    }

    // in-wave LDS transpose round-trip: DS ops are in-order, wait then reload
    asm volatile("s_wait_dscnt 0" ::: "memory");

    v16bf pa;
    {
      const __bf16* pr = &pbuf[wslot][l15 * 32 + half * 8];
      v8bf lo = *(const v8bf*)pr;
      v8bf hi = *(const v8bf*)(pr + 16);
#pragma unroll
      for (int i = 0; i < 8; ++i) { pa[i] = lo[i]; pa[i + 8] = hi[i]; }
    }

    // ---- O += P x V (zero-padded V^T makes edge columns contribute 0) ----
#pragma unroll
    for (int n = 0; n < 16; ++n) {
      const __bf16* vr = vtb + ((size_t)(bb * EDIM + n * 16 + l15)) * SP
                             + (TOFF + t0) + half * 8;
      v16bf vb = load_frag(vr);
      o[n] = __builtin_amdgcn_wmma_f32_16x16x32_bf16(false, pa, false, vb,
                                                     (short)0, o[n], false, false);
    }
  }

  // ---- normalize and store f32 output ----
#pragma unroll
  for (int j = 0; j < 8; ++j) {
    float inv = 1.0f / ssum[j];
    int s = S0 + half * 8 + j;
    float* orow = out + ((size_t)(bb * SEQ + s)) * EDIM + l15;
#pragma unroll
    for (int n = 0; n < 16; ++n) orow[n * 16] = o[n][j] * inv;
  }
}

extern "C" void kernel_launch(void* const* d_in, const int* in_sizes, int n_in,
                              void* d_out, int out_size, void* d_ws, size_t ws_size,
                              hipStream_t stream) {
  const float* x  = (const float*)d_in[0];
  const float* Wq = (const float*)d_in[1];
  const float* bq = (const float*)d_in[2];
  const float* Wk = (const float*)d_in[3];
  const float* bk = (const float*)d_in[4];
  const float* Wv = (const float*)d_in[5];
  const float* bv = (const float*)d_in[6];

  char* ws = (char*)d_ws;
  __bf16* xb   = (__bf16*)(ws);                        // 2 MB  x bf16
  __bf16* wb   = (__bf16*)(ws + (size_t)(2u << 20));   // 384 KB Wq|Wk|Wv bf16
  __bf16* qb   = (__bf16*)(ws + (size_t)(3u << 20));   // 2 MB  q bf16
  __bf16* kbuf = (__bf16*)(ws + (size_t)(5u << 20));   // ~2.4 MB padded K bf16
  __bf16* vtb  = (__bf16*)(ws + (size_t)(8u << 20));   // ~2.4 MB padded V^T bf16

  const int npad = BATCH * SP * EDIM;                  // bf16 elements (even)
  zero_ws<<<256, 256, 0, stream>>>((uint32_t*)kbuf, npad / 2);
  zero_ws<<<256, 256, 0, stream>>>((uint32_t*)vtb,  npad / 2);

  cvt_bf16<<<512, 256, 0, stream>>>(x,  xb, BATCH * SEQ * EDIM);
  cvt_bf16<<<64, 256, 0, stream>>>(Wq, wb,                   EDIM * EDIM);
  cvt_bf16<<<64, 256, 0, stream>>>(Wk, wb +     EDIM * EDIM, EDIM * EDIM);
  cvt_bf16<<<64, 256, 0, stream>>>(Wv, wb + 2 * EDIM * EDIM, EDIM * EDIM);

  qkv_proj<<<1536, 256, 0, stream>>>(xb, wb, bq, bk, bv, qb, kbuf, vtb);
  local_attn<<<32, 256, 0, stream>>>(qb, kbuf, vtb, (float*)d_out);
}
